// DynamicSlotsSoftMoE_2740189135144
// MI455X (gfx1250) — matrix operand order repaired and checked by reading, not compiled, not run on hardware
//
#include <hip/hip_runtime.h>
#include <math.h>

// ---------------------------------------------------------------------------
// CDNA5 (gfx1250) types / feature probes
// ---------------------------------------------------------------------------
typedef __attribute__((ext_vector_type(16))) __bf16 v16bf;
typedef __attribute__((ext_vector_type(8)))  float  v8f;
typedef __attribute__((ext_vector_type(4)))  unsigned int v4u;
typedef __attribute__((ext_vector_type(8)))  int    v8i;
typedef __attribute__((ext_vector_type(4)))  int    v4i;

#if defined(__has_builtin)
#  if __has_builtin(__builtin_amdgcn_tensor_load_to_lds) && \
      __has_builtin(__builtin_amdgcn_s_wait_tensorcnt)
#    define HAS_TDM 1
#  endif
#endif
#ifndef HAS_TDM
#  define HAS_TDM 0
#endif

#define BM 128
#define BN 128
#define BK 32
#define LDST 40   // LDS row stride in bf16 elements (80B = 64B row + 16B pad)

__device__ __forceinline__ unsigned short f2bf(float f) {
    unsigned u = __float_as_uint(f);
    u += 0x7FFFu + ((u >> 16) & 1u);      // round-to-nearest-even
    return (unsigned short)(u >> 16);
}

union Pack4 { unsigned short s[4]; uint2 u; };
union FragU { uint4 q[2]; v16bf v; };

__device__ __forceinline__ void load4cvt(const float* p, Pack4& o) {
    float4 v = *(const float4*)p;
    o.s[0] = f2bf(v.x); o.s[1] = f2bf(v.y); o.s[2] = f2bf(v.z); o.s[3] = f2bf(v.w);
}
__device__ __forceinline__ void load4cvt(const unsigned short* p, Pack4& o) {
    o.u = *(const uint2*)p;
}

// ---------------------------------------------------------------------------
// Tensor Data Mover: DMA a 128-row x 32-element bf16 tile (row stride ldElems)
// from global into LDS with a padded row stride of 40 elements (80 bytes).
// D# per CDNA5 ISA ch.8: pad_interval=3 (pad after every 16 DWORDs = 64B),
// pad_amount=3 (4 DWORDs = 16B)  ->  LDS row pitch 80B.
// This toolchain exposes the 6-arg builtin:
//   (uint32x4 g0, int32x8 g1, int32x4, int32x4, int32x8, i32 cpol)
// ---------------------------------------------------------------------------
__device__ __forceinline__ void tdm_stage(const unsigned short* gtile, int ldElems,
                                          unsigned short* lds)
{
#if HAS_TDM
    unsigned long long ga = (unsigned long long)(const void*)gtile;
    unsigned ldsAddr = (unsigned)(unsigned long long)(void*)lds;  // LDS aperture: offset in [31:0]
    v4u g0;
    g0.x = 1u;                                                // count=1, user mode
    g0.y = ldsAddr;                                           // lds_addr
    g0.z = (unsigned)(ga & 0xFFFFFFFFull);                    // global_addr[31:0]
    g0.w = (unsigned)((ga >> 32) & 0x01FFFFFFull) | 0x80000000u; // addr[56:32] | type=2
    v8i g1;
    g1[0] = (int)((1u << 16) | (1u << 20) | (3u << 22) | (3u << 25)); // data_size=2B, pad_en, int=16DW, amt=4DW
    g1[1] = 0;                                                // barrier addr 0, tensor_dim0[15:0]=0
    g1[2] = 0x4000;                                           // tensor_dim0 = 1<<30 (no clip)
    g1[3] = (int)(0x4000u | (32u << 16));                     // tensor_dim1 = 1<<30, tile_dim0 = 32
    g1[4] = 128;                                              // tile_dim1 = 128 (tile_dim2 = 0)
    g1[5] = ldElems;                                          // tensor_dim0_stride[31:0]
    g1[6] = 0;
    g1[7] = 0;
    v4i gz4 = {0, 0, 0, 0};
    v8i gz8 = {0, 0, 0, 0, 0, 0, 0, 0};
    __builtin_amdgcn_tensor_load_to_lds(g0, g1, gz4, gz4, gz8, 0);
#else
    (void)gtile; (void)ldElems; (void)lds;
#endif
}

__device__ __forceinline__ void tdm_wait()
{
#if HAS_TDM
    __builtin_amdgcn_s_wait_tensorcnt(0);
#endif
}

// ---------------------------------------------------------------------------
// Staging helpers (convert fp32 -> bf16, or raw bf16 copy fallback)
// ---------------------------------------------------------------------------
template<typename T>
__device__ __forceinline__ void stage_rows(const T* base, int ld, int rowBase, int kk,
                                           unsigned short* lds, int tid)
{
    #pragma unroll
    for (int it = 0; it < 4; ++it) {
        int lin = (it * 256 + tid) * 4;
        int row = lin >> 5, col = lin & 31;
        const T* p = base + (size_t)(rowBase + row) * ld + (kk + col);
        Pack4 pk; load4cvt(p, pk);
        __builtin_prefetch(p + BK, 0, 1);      // global_prefetch next K tile
        *(uint2*)&lds[row * LDST + col] = pk.u;
    }
}

template<typename T>
__device__ __forceinline__ void stage_transpose(const T* base, int ld, int nBase, int kk,
                                                unsigned short* lds, int tid)
{
    #pragma unroll
    for (int it = 0; it < 4; ++it) {
        int lin = (it * 256 + tid) * 4;
        int k = lin >> 7, n0 = lin & 127;
        const T* p = base + (size_t)(kk + k) * ld + (nBase + n0);
        Pack4 pk; load4cvt(p, pk);
        __builtin_prefetch(p + (size_t)BK * ld, 0, 1);
        lds[(n0 + 0) * LDST + k] = pk.s[0];
        lds[(n0 + 1) * LDST + k] = pk.s[1];
        lds[(n0 + 2) * LDST + k] = pk.s[2];
        lds[(n0 + 3) * LDST + k] = pk.s[3];
    }
}

// ---------------------------------------------------------------------------
// Templated bf16 WMMA GEMM, fp32 accumulate, double-buffered LDS + TDM.
//   C[z] = epilogue( A[z/adiv] (M x K) * op(B[z%bmod]) (K x N) )
//   TRANSB=1 : B stored [N][K] row-major (NT). TRANSB=0 : B stored [K][N] (NN).
//   EPI: 0 = f32 store   1 = bf16 (scaled)   2 = bf16 scale+bias+GELU   3 = bf16 scale+bias
// ---------------------------------------------------------------------------
template<int TRANSB, int EPI, typename TA, typename TB>
__global__ __launch_bounds__(256)
void gemm_wmma(const TA* __restrict__ Ag, int lda, long long strideA, int adiv,
               const TB* __restrict__ Bg, int ldb, long long strideB, int bmod,
               void* __restrict__ Cg, int ldc, long long strideC,
               const float* __restrict__ scale, int smod,
               const float* __restrict__ bias, int biasStride,
               int K)
{
    __shared__ __align__(16) unsigned short ldsA[2 * BM * LDST];
    __shared__ __align__(16) unsigned short ldsB[2 * BN * LDST];

    constexpr bool TDM_A = HAS_TDM && (sizeof(TA) == 2);
    constexpr bool TDM_B = HAS_TDM && TRANSB && (sizeof(TB) == 2);
    constexpr bool TDM_ANY = TDM_A || TDM_B;

    const int z     = blockIdx.z;
    const int tid   = threadIdx.x;
    const int mBase = blockIdx.y * BM;
    const int nBase = blockIdx.x * BN;

    const TA* A = Ag + (size_t)(z / adiv) * (size_t)strideA;
    const TB* B = Bg + (size_t)(z % bmod) * (size_t)strideB;
    const float sc = scale ? scale[z % smod] : 1.0f;
    const float* bp = bias ? (bias + (size_t)(z % bmod) * biasStride) : nullptr;

    const int lane = tid & 31;
    const int wave = tid >> 5;
    const int wm = wave & 1;     // 2 waves along M (64-row slabs)
    const int wn = wave >> 1;    // 4 waves along N (32-col slabs)
    const int hh = lane >> 4;    // lane half
    const int r  = lane & 15;

    const v8f zv = {0.f,0.f,0.f,0.f,0.f,0.f,0.f,0.f};
    v8f acc[4][2];
    #pragma unroll
    for (int t = 0; t < 4; ++t)
        #pragma unroll
        for (int u = 0; u < 2; ++u) acc[t][u] = zv;

    auto stage_all = [&](int kk, int bi) {
        unsigned short* bufA = &ldsA[bi * (BM * LDST)];
        unsigned short* bufB = &ldsB[bi * (BN * LDST)];
        if constexpr (TDM_A) {
            if (wave == 0)
                tdm_stage((const unsigned short*)A + (size_t)mBase * lda + kk, lda, bufA);
        } else {
            stage_rows(A, lda, mBase, kk, bufA, tid);
        }
        if constexpr (TRANSB) {
            if constexpr (TDM_B) {
                if (wave == 0)
                    tdm_stage((const unsigned short*)B + (size_t)nBase * ldb + kk, ldb, bufB);
            } else {
                stage_rows(B, ldb, nBase, kk, bufB, tid);
            }
        } else {
            stage_transpose(B, ldb, nBase, kk, bufB, tid);
        }
    };
    auto stage_fence = [&]() {
        if (TDM_ANY && wave == 0) tdm_wait();
        __syncthreads();
    };

    const int steps = K / BK;
    stage_all(0, 0);
    stage_fence();

    for (int s = 0; s < steps; ++s) {
        const int cur = s & 1;
        if (s + 1 < steps) stage_all((s + 1) * BK, cur ^ 1);

        const unsigned short* baseA = &ldsA[cur * (BM * LDST)];
        const unsigned short* baseB = &ldsB[cur * (BN * LDST)];
        FragU fa[4], fb[2];
        #pragma unroll
        for (int t = 0; t < 4; ++t) {
            const unsigned short* rp = &baseA[(wm * 64 + t * 16 + r) * LDST];
            fa[t].q[0] = *(const uint4*)(rp + hh * 8);        // K = h*8 .. +7
            fa[t].q[1] = *(const uint4*)(rp + 16 + hh * 8);   // K = 16+h*8 .. +7
        }
        #pragma unroll
        for (int u = 0; u < 2; ++u) {
            const unsigned short* rp = &baseB[(wn * 32 + u * 16 + r) * LDST];
            fb[u].q[0] = *(const uint4*)(rp + hh * 8);        // K = h*8 .. +7
            fb[u].q[1] = *(const uint4*)(rp + 16 + hh * 8);   // K = 16+h*8 .. +7
        }
        #pragma unroll
        for (int t = 0; t < 4; ++t)
            #pragma unroll
            for (int u = 0; u < 2; ++u)
                acc[t][u] = __builtin_amdgcn_wmma_f32_16x16x32_bf16(
                    false, fa[t].v, false, fb[u].v, (short)0, acc[t][u], false, false);

        stage_fence();
    }

    // ---- epilogue (C/D layout: VGPR i -> M = h*8+i, N = lane&15) ----
    char* Cb = (char*)Cg;
    #pragma unroll
    for (int t = 0; t < 4; ++t) {
        int rowBase = mBase + wm * 64 + t * 16 + hh * 8;
        #pragma unroll
        for (int u = 0; u < 2; ++u) {
            int col = nBase + wn * 32 + u * 16 + r;
            float bv = bp ? bp[col] : 0.0f;
            #pragma unroll
            for (int i = 0; i < 8; ++i) {
                float v = acc[t][u][i] * sc + bv;
                size_t idx = (size_t)z * (size_t)strideC + (size_t)(rowBase + i) * ldc + col;
                if (EPI == 0) {
                    ((float*)Cb)[idx] = v;
                } else if (EPI == 1) {
                    ((unsigned short*)Cb)[idx] = f2bf(v);
                } else if (EPI == 2) {
                    float g = 0.5f * v * (1.0f + erff(v * 0.70710678118654752f));
                    ((unsigned short*)Cb)[idx] = f2bf(g);
                } else {
                    ((unsigned short*)Cb)[idx] = f2bf(v);
                }
            }
        }
    }
}

// ---------------------------------------------------------------------------
// Spectral-norm scalar helpers:  scale = sigma * ||t|| / ||W^T t||,  t = W u
// ---------------------------------------------------------------------------
__global__ void matvec_rows(const float* __restrict__ W, const float* __restrict__ u,
                            int C, float* __restrict__ t, float* __restrict__ nrm)
{
    __shared__ float red[256];
    int row = blockIdx.x;
    const float* Wr = W + (size_t)row * C;
    float p = 0.f;
    for (int c = threadIdx.x; c < C; c += 256) p += Wr[c] * u[c];
    red[threadIdx.x] = p; __syncthreads();
    for (int s = 128; s > 0; s >>= 1) {
        if (threadIdx.x < s) red[threadIdx.x] += red[threadIdx.x + s];
        __syncthreads();
    }
    if (threadIdx.x == 0) { float v = red[0]; t[row] = v; atomicAdd(nrm, v * v); }
}

__global__ void matvec_cols(const float* __restrict__ W, const float* __restrict__ t,
                            int R, int C, float* __restrict__ nrm)
{
    __shared__ float red[256];
    int c  = blockIdx.x * 64 + (threadIdx.x & 63);
    int r0 = threadIdx.x >> 6;
    float p = 0.f;
    for (int r = r0; r < R; r += 4) p += W[(size_t)r * C + c] * t[r];
    red[threadIdx.x] = p; __syncthreads();
    if (threadIdx.x < 64) {
        float g = red[threadIdx.x] + red[threadIdx.x + 64] +
                  red[threadIdx.x + 128] + red[threadIdx.x + 192];
        atomicAdd(nrm, g * g);
    }
}

__global__ void finalize_scales(const float* __restrict__ nrmT, const float* __restrict__ nrmG,
                                const float* __restrict__ sigma_slot,
                                const float* __restrict__ sig1, const float* __restrict__ sig2,
                                float* __restrict__ scales)
{
    int i = threadIdx.x;
    if (i < 9) {
        float sg = (i == 0) ? sigma_slot[0] : (i < 5 ? sig1[i - 1] : sig2[i - 5]);
        scales[i] = sg * sqrtf(nrmT[i] / nrmG[i]);
    }
}

// xm[b,s,d] = mean over e of x[b, s*4+e, d]
__global__ void mean_kernel(const float* __restrict__ x, float* __restrict__ xm)
{
    int bs = blockIdx.x;                      // b*512 + sIdx
    size_t base = (size_t)bs * 4096;
    int d0 = threadIdx.x * 4;
    float4 a0 = *(const float4*)(x + base + 0 * 1024 + d0);
    float4 a1 = *(const float4*)(x + base + 1 * 1024 + d0);
    float4 a2 = *(const float4*)(x + base + 2 * 1024 + d0);
    float4 a3 = *(const float4*)(x + base + 3 * 1024 + d0);
    float4 o;
    o.x = 0.25f * (a0.x + a1.x + a2.x + a3.x);
    o.y = 0.25f * (a0.y + a1.y + a2.y + a3.y);
    o.z = 0.25f * (a0.z + a1.z + a2.z + a3.z);
    o.w = 0.25f * (a0.w + a1.w + a2.w + a3.w);
    *(float4*)(xm + (size_t)bs * 1024 + d0) = o;
}

// combine = softmax over last dim of L[b,n,:2048] -> bf16
__global__ void softmax_rows(const float* __restrict__ L, unsigned short* __restrict__ O)
{
    __shared__ float red[256];
    __shared__ float mS, sS;
    size_t base = (size_t)blockIdx.x * 2048;
    float m = -3.4e38f;
    for (int i = threadIdx.x; i < 2048; i += 256) m = fmaxf(m, L[base + i]);
    red[threadIdx.x] = m; __syncthreads();
    for (int s = 128; s > 0; s >>= 1) {
        if (threadIdx.x < s) red[threadIdx.x] = fmaxf(red[threadIdx.x], red[threadIdx.x + s]);
        __syncthreads();
    }
    if (threadIdx.x == 0) mS = red[0];
    __syncthreads();
    float mm = mS, p = 0.f;
    for (int i = threadIdx.x; i < 2048; i += 256) p += __expf(L[base + i] - mm);
    red[threadIdx.x] = p; __syncthreads();
    for (int s = 128; s > 0; s >>= 1) {
        if (threadIdx.x < s) red[threadIdx.x] += red[threadIdx.x + s];
        __syncthreads();
    }
    if (threadIdx.x == 0) sS = 1.0f / red[0];
    __syncthreads();
    float inv = sS;
    for (int i = threadIdx.x; i < 2048; i += 256)
        O[base + i] = f2bf(__expf(L[base + i] - mm) * inv);
}

// dispatch^T[b,es,n] = softmax over n of L[b,n,es] -> bf16 (transposed store)
__global__ void softmax_cols(const float* __restrict__ L, unsigned short* __restrict__ O)
{
    __shared__ float red[256];
    __shared__ float mS, sS;
    int b  = blockIdx.x >> 11;
    int es = blockIdx.x & 2047;
    const float* Lb = L + (size_t)b * 2048 * 2048 + es;
    float m = -3.4e38f;
    for (int n = threadIdx.x; n < 2048; n += 256) m = fmaxf(m, Lb[(size_t)n * 2048]);
    red[threadIdx.x] = m; __syncthreads();
    for (int s = 128; s > 0; s >>= 1) {
        if (threadIdx.x < s) red[threadIdx.x] = fmaxf(red[threadIdx.x], red[threadIdx.x + s]);
        __syncthreads();
    }
    if (threadIdx.x == 0) mS = red[0];
    __syncthreads();
    float mm = mS, p = 0.f;
    for (int n = threadIdx.x; n < 2048; n += 256) p += __expf(Lb[(size_t)n * 2048] - mm);
    red[threadIdx.x] = p; __syncthreads();
    for (int s = 128; s > 0; s >>= 1) {
        if (threadIdx.x < s) red[threadIdx.x] += red[threadIdx.x + s];
        __syncthreads();
    }
    if (threadIdx.x == 0) sS = 1.0f / red[0];
    __syncthreads();
    float inv = sS;
    size_t obase = (size_t)blockIdx.x * 2048;
    for (int n = threadIdx.x; n < 2048; n += 256)
        O[obase + n] = f2bf(__expf(Lb[(size_t)n * 2048] - mm) * inv);
}

// ---------------------------------------------------------------------------
extern "C" void kernel_launch(void* const* d_in, const int* in_sizes, int n_in,
                              void* d_out, int out_size, void* d_ws, size_t ws_size,
                              hipStream_t stream)
{
    const float* x          = (const float*)d_in[0];
    const float* W_slot     = (const float*)d_in[1];
    const float* u_slot     = (const float*)d_in[2];
    const float* sigma_slot = (const float*)d_in[3];
    const float* W1         = (const float*)d_in[4];
    const float* b1         = (const float*)d_in[5];
    const float* u1         = (const float*)d_in[6];
    const float* sig1       = (const float*)d_in[7];
    const float* W2         = (const float*)d_in[8];
    const float* b2         = (const float*)d_in[9];
    const float* u2         = (const float*)d_in[10];
    const float* sig2       = (const float*)d_in[11];
    float* out = (float*)d_out;
    (void)in_sizes; (void)n_in; (void)out_size; (void)ws_size;

    const int Bb = 4, N = 2048, D = 1024, H = 4096, S = 512;

    char* ws = (char*)d_ws;
    size_t off = 0;
    auto alloc = [&](size_t bytes) {
        size_t o = off; off += (bytes + 255) & ~(size_t)255; return o;
    };
    float* scal   = (float*)(ws + alloc(1024));
    float* nrmT   = scal + 0;   // [9]
    float* nrmG   = scal + 16;  // [9]
    float* scales = scal + 32;  // [9]
    float* tbuf   = (float*)(ws + alloc(4096 * sizeof(float)));
    float* xm     = (float*)(ws + alloc((size_t)Bb * S * D * 4));
    unsigned short* SEbf = (unsigned short*)(ws + alloc((size_t)Bb * N * D * 2)); // -> Ybf later
    float* L      = (float*)(ws + alloc((size_t)Bb * N * N * 4));                 // -> Hbf later
    unsigned short* Cbf  = (unsigned short*)(ws + alloc((size_t)Bb * N * N * 2));
    unsigned short* Dtbf = (unsigned short*)(ws + alloc((size_t)Bb * N * N * 2));
    unsigned short* Slbf = (unsigned short*)(ws + alloc((size_t)Bb * N * D * 2));
    unsigned short* Hbf  = (unsigned short*)L;     // lifetime-disjoint alias
    unsigned short* Ybf  = SEbf;                   // lifetime-disjoint alias

    (void)hipMemsetAsync(scal, 0, 1024, stream);

    // --- spectral-norm scales (9 matrices) ---
    matvec_rows<<<4096, 256, 0, stream>>>(W_slot, u_slot, 1024, tbuf, nrmT + 0);
    matvec_cols<<<1024 / 64, 256, 0, stream>>>(W_slot, tbuf, 4096, 1024, nrmG + 0);
    for (int e = 0; e < 4; ++e) {
        matvec_rows<<<4096, 256, 0, stream>>>(W1 + (size_t)e * H * D, u1 + (size_t)e * D,
                                              1024, tbuf, nrmT + 1 + e);
        matvec_cols<<<1024 / 64, 256, 0, stream>>>(W1 + (size_t)e * H * D, tbuf,
                                                   4096, 1024, nrmG + 1 + e);
    }
    for (int e = 0; e < 4; ++e) {
        matvec_rows<<<1024, 256, 0, stream>>>(W2 + (size_t)e * D * H, u2 + (size_t)e * H,
                                              4096, tbuf, nrmT + 5 + e);
        matvec_cols<<<4096 / 64, 256, 0, stream>>>(W2 + (size_t)e * D * H, tbuf,
                                                   1024, 4096, nrmG + 5 + e);
    }
    finalize_scales<<<1, 16, 0, stream>>>(nrmT, nrmG, sigma_slot, sig1, sig2, scales);

    // --- segment means ---
    mean_kernel<<<Bb * S, 256, 0, stream>>>(x, xm);

    // (a) slot_embeds[b,e*512+s,d] = scale0 * (xm[b] @ Wsl_e^T)   -> bf16
    gemm_wmma<1, 1, float, float><<<dim3(D / 128, S / 128, 16), 256, 0, stream>>>(
        xm, D, (long long)S * D, 4,
        W_slot, D, (long long)D * D, 4,
        SEbf, D, (long long)S * D,
        scales + 0, 1, nullptr, 0, D);

    // (b) logits[b,n,es] = x[b] @ SE[b]^T   -> f32   (B staged via TDM)
    gemm_wmma<1, 0, float, unsigned short><<<dim3(N / 128, N / 128, 4), 256, 0, stream>>>(
        x, D, (long long)N * D, 1,
        SEbf, D, (long long)N * D, 1 << 30,
        L, N, (long long)N * N,
        nullptr, 1, nullptr, 0, D);

    // (c,d) softmaxes
    softmax_rows<<<Bb * N, 256, 0, stream>>>(L, Cbf);
    softmax_cols<<<Bb * N, 256, 0, stream>>>(L, Dtbf);

    // (e) slots[b,es,d] = dispatch^T[b] @ x[b]   (NN, A via TDM) -> bf16
    gemm_wmma<0, 1, unsigned short, float><<<dim3(D / 128, N / 128, 4), 256, 0, stream>>>(
        Dtbf, N, (long long)N * N, 1,
        x, D, (long long)N * D, 1 << 30,
        Slbf, D, (long long)N * D,
        nullptr, 1, nullptr, 0, N);

    // (f) hidden = gelu(scale1[e]*(slots @ W1_e^T) + b1[e])  (A via TDM) -> bf16
    gemm_wmma<1, 2, unsigned short, float><<<dim3(H / 128, S / 128, 16), 256, 0, stream>>>(
        Slbf, D, (long long)S * D, 1,
        W1, D, (long long)H * D, 4,
        Hbf, H, (long long)S * H,
        scales + 1, 4, b1, H, D);

    // (g) y = scale2[e]*(hidden @ W2_e^T) + b2[e]  (A via TDM) -> bf16
    gemm_wmma<1, 3, unsigned short, float><<<dim3(D / 128, S / 128, 16), 256, 0, stream>>>(
        Hbf, H, (long long)S * H, 1,
        W2, H, (long long)D * H, 4,
        Ybf, D, (long long)S * D,
        scales + 5, 4, b2, D, H);

    // (h) out[b,n,d] = combine[b] @ y[b]  (NN, A via TDM) -> f32
    gemm_wmma<0, 0, unsigned short, unsigned short><<<dim3(D / 128, N / 128, 4), 256, 0, stream>>>(
        Cbf, N, (long long)N * N, 1,
        Ybf, D, (long long)N * D, 1 << 30,
        out, D, (long long)N * D,
        nullptr, 1, nullptr, 0, N);
}
